// selfBank_31215822308175
// MI455X (gfx1250) — compile-verified, matrix operand order checked by reading, not run on hardware
//
#include <hip/hip_runtime.h>
#include <hip/hip_bf16.h>
#include <math.h>

// ---------------------------------------------------------------------------
// Mel filterbank log-energies on MI455X (gfx1250, wave32, WMMA).
// Stage 1: Re/Im = Frames(16x25->16x32 f16) x Twiddle(32x16 f16)  [v_wmma_f32_16x16x32_f16]
// Stage 2: FB   += Pow(16x16 f32) x W(16x16 f32) in K=4 steps     [v_wmma_f32_16x16x4_f32]
//          banded: zero mel-tile blocks skipped via uniform mask branch.
// Signal window staged via GLOBAL_LOAD_ASYNC_TO_LDS_B32 (ASYNCcnt path).
// ---------------------------------------------------------------------------

typedef __attribute__((ext_vector_type(16))) _Float16 v16h;
typedef __attribute__((ext_vector_type(8)))  float    v8f;
typedef __attribute__((ext_vector_type(2)))  float    v2f;

#define NFFT       512
#define NBINS      257     // NFFT/2+1
#define NFILT      40
#define FRAME_LEN  25      // round(0.025*1000)
#define FRAME_STEP 10      // round(0.010*1000)
#define NBATCH     32
#define SIGLEN     50000
#define NFRAMES    4999    // ceil((50000-25)/10)+1
#define FTILES     17      // ceil(257/16)
#define MTILES     3       // ceil(40/16)
#define FRTILES    313     // ceil(4999/16)
#define WAVES_PB   8
#define EPSF       2.2204460492503131e-16f
#define TWO_PI     6.283185307179586476925286766559

#define USE_ASYNC_LDS 1    // gfx1250 GLOBAL_LOAD_ASYNC_TO_LDS_B32 staging

// d_ws layout: [twiddle f16][align 256][Wb f32][band masks int]
#define TW_HALFS  (FTILES * 2 * 512)                    // 17408 halves
#define WB_OFF    (((TW_HALFS * 2) + 255) & ~255)       // byte offset of Wb
#define WB_FLOATS (MTILES * FTILES * 4 * 64)            // 13056 floats
#define MASK_OFF  (WB_OFF + WB_FLOATS * 4)              // 17 ints

// ---------------------------------------------------------------------------
// Setup kernel 1: cos/sin twiddles in exact 16-bit B-matrix register layout;
// also zeroes the mel-band masks (filled by fbank_init via atomicOr).
// B[K][n]: lane -> n = lane&15, kbase = (lane<16 ? 0 : 8);
//          e<8 -> K = kbase+e ; e>=8 -> K = kbase+8+e.
// ---------------------------------------------------------------------------
__global__ void twiddle_init(_Float16* __restrict__ tw, int* __restrict__ wmask) {
  int idx = blockIdx.x * 256 + threadIdx.x;
  if (blockIdx.x == 0 && threadIdx.x < FTILES) wmask[threadIdx.x] = 0;
  if (idx >= TW_HALFS) return;
  int tile = idx >> 9;         // /512
  int rem  = idx & 511;
  int lane = rem >> 4;
  int e    = rem & 15;
  int ft   = tile >> 1;
  int cs   = tile & 1;
  int nf   = ft * 16 + (lane & 15);
  int kbase = (lane < 16) ? 0 : 8;
  int K     = kbase + ((e < 8) ? e : (8 + e));
  float v = 0.0f;
  if (K < FRAME_LEN && nf < NBINS) {
    int ph = (K * nf) & (NFFT - 1);                 // exact phase mod 512
    double ang = (TWO_PI / (double)NFFT) * (double)ph;
    v = (float)(cs ? sin(ang) : cos(ang));
  }
  tw[idx] = (_Float16)v;
}

// ---------------------------------------------------------------------------
// Setup kernel 2: mel filterbank weights in 32-bit B-matrix register layout
// for V_WMMA_F32_16X16X4_F32 ([mt][ft][kk][lane][e2], 8B/lane coalesced),
// plus per-freq-tile band mask: bit mt set iff W block (ft,mt) has nonzeros.
// Filterbank math mirrors the NumPy reference in float64.
// ---------------------------------------------------------------------------
__global__ void fbank_init(float* __restrict__ wb, int* __restrict__ wmask,
                           const int* __restrict__ fsp) {
  int idx = blockIdx.x * 256 + threadIdx.x;
  if (idx >= WB_FLOATS) return;
  int e2   = idx & 1;
  int lane = (idx >> 1) & 31;
  int kk   = (idx >> 6) & 3;
  int t    = idx >> 8;               // mt*FTILES + ft
  int ft   = t % FTILES;
  int mt   = t / FTILES;
  int kloc = kk * 4 + ((lane < 16) ? 0 : 2) + e2;
  int f    = ft * 16 + kloc;         // global freq bin
  int mel  = mt * 16 + (lane & 15);  // filter index
  float val = 0.0f;
  if (f < NBINS && mel < NFILT && f >= 1) {
    double fs  = (double)fsp[0];
    double hfm = 2595.0 * log10(1.0 + fs / 2.0 / 700.0);
    auto bpt = [&](int i) -> double {
      double mp = hfm * (double)i / (double)(NFILT + 1);   // linspace, 42 pts
      double hz = 700.0 * (pow(10.0, mp / 2595.0) - 1.0);
      return hz / (fs / 2.0) * ((double)NFFT / 2.0);
    };
    double b0 = bpt(mel), b1 = bpt(mel + 1), b2 = bpt(mel + 2);
    int left = (int)b0, center = (int)b1, right = (int)b2;
    int j = f - 1;
    if (j >= left && j < center)       val = (float)(((double)f - b0) / (b1 - b0));
    else if (j >= center && j < right) val = (float)((b2 - (double)f) / (b2 - b1));
  }
  wb[idx] = val;
  if (val != 0.0f) atomicOr(&wmask[ft], 1 << mt);
}

// ---------------------------------------------------------------------------
// Main fused kernel: one wave per 16-frame tile.
// ---------------------------------------------------------------------------
__global__ __launch_bounds__(256) void mel_wmma(const float* __restrict__ sig,
                                                const _Float16* __restrict__ tw,
                                                const float* __restrict__ wb,
                                                const int* __restrict__ wmask,
                                                float* __restrict__ out) {
  __shared__ float lds_sig[WAVES_PB][176 + 4];   // contiguous signal window
  __shared__ float lds_pow[WAVES_PB][16 * 17];   // pow tile, padded stride 17

  const int lane = threadIdx.x & 31;
  const int wid  = threadIdx.x >> 5;
  const int gw   = blockIdx.x * WAVES_PB + wid;
  const int b    = gw / FRTILES;
  const int frt  = gw % FRTILES;
  if (b >= NBATCH) return;
  const int frbase = frt * 16;
  const int sbase  = frbase * FRAME_STEP;

  // ---- stage-in: 176 contiguous floats -> LDS (async path, EXEC-masked) ----
  const float* sigb = sig + (size_t)b * SIGLEN;
  for (int ti = lane; ti < 176; ti += 32) {
    int idx = sbase + ti;
    if (idx < SIGLEN) {
#if USE_ASYNC_LDS
      unsigned lds_off =
          (unsigned)(uintptr_t)(__attribute__((address_space(3))) float*)&lds_sig[wid][ti];
      unsigned long long ga = (unsigned long long)(uintptr_t)(sigb + idx);
      asm volatile("global_load_async_to_lds_b32 %0, %1, off"
                   :: "v"(lds_off), "v"(ga) : "memory");
#else
      lds_sig[wid][ti] = sigb[idx];
#endif
    } else {
      lds_sig[wid][ti] = 0.0f;       // reference zero-pads the tail
    }
  }
#if USE_ASYNC_LDS
  asm volatile("s_wait_asynccnt 0x0" ::: "memory");
#endif
  __builtin_amdgcn_wave_barrier();

  // ---- build A: 16x32 f16 frame tile in A-register layout ----
  const int row   = lane & 15;
  const int kbase = (lane < 16) ? 0 : 8;
  const bool rowok = (frbase + row) < NFRAMES;
  v16h a;
#pragma unroll
  for (int e = 0; e < 16; ++e) {
    int K = kbase + ((e < 8) ? e : (8 + e));
    float v = (rowok && K < FRAME_LEN) ? lds_sig[wid][row * FRAME_STEP + K] : 0.0f;
    a[e] = (_Float16)v;
  }

  v8f acc0 = {}, acc1 = {}, acc2 = {};   // FB accumulators, mel tiles 0..2

  const int mofs = (lane < 16) ? 0 : 8;  // C/D layout row offset
  const int col  = lane & 15;            // C/D layout column
  const int m    = lane & 15;            // stage-2 A row
  const int ksel = (lane < 16) ? 0 : 2;  // stage-2 A K select

  for (int ft = 0; ft < FTILES; ++ft) {
    // twiddle B tiles: one v16h (32B) per lane, coalesced, L2-resident
    v16h bcos = ((const v16h*)(tw + (size_t)(ft * 2 + 0) * 512))[lane];
    v16h bsin = ((const v16h*)(tw + (size_t)(ft * 2 + 1) * 512))[lane];

    v8f zero = {};
    v8f re = __builtin_amdgcn_wmma_f32_16x16x32_f16(false, a, false, bcos,
                                                    (short)0, zero, false, false);
    v8f im = __builtin_amdgcn_wmma_f32_16x16x32_f16(false, a, false, bsin,
                                                    (short)0, zero, false, false);
    v8f p = (re * re + im * im) * (1.0f / (float)NFFT);

    // C/D layout -> plain [M][K] in LDS (intra-wave transpose)
#pragma unroll
    for (int r = 0; r < 8; ++r)
      lds_pow[wid][(mofs + r) * 17 + col] = p[r];
    __builtin_amdgcn_wave_barrier();

    // A2 fragments for all 4 K=4 steps (shared by every mel tile)
    v2f a2[4];
#pragma unroll
    for (int kk = 0; kk < 4; ++kk) {
      a2[kk].x = lds_pow[wid][m * 17 + kk * 4 + ksel + 0];
      a2[kk].y = lds_pow[wid][m * 17 + kk * 4 + ksel + 1];
    }

    // banded stage 2: skip all-zero W blocks (uniform scalar branch; EXEC
    // stays all-ones so WMMA remains legal inside the branches)
    const int msk = __builtin_amdgcn_readfirstlane(wmask[ft]);
    if (msk & 1) {
#pragma unroll
      for (int kk = 0; kk < 4; ++kk) {
        const float* p0 = wb + (size_t)(((0 * FTILES + ft) * 4 + kk) * 64) + lane * 2;
        v2f b2; b2.x = p0[0]; b2.y = p0[1];
        acc0 = __builtin_amdgcn_wmma_f32_16x16x4_f32(false, a2[kk], false, b2,
                                                     (short)0, acc0, false, false);
      }
    }
    if (msk & 2) {
#pragma unroll
      for (int kk = 0; kk < 4; ++kk) {
        const float* p1 = wb + (size_t)(((1 * FTILES + ft) * 4 + kk) * 64) + lane * 2;
        v2f b2; b2.x = p1[0]; b2.y = p1[1];
        acc1 = __builtin_amdgcn_wmma_f32_16x16x4_f32(false, a2[kk], false, b2,
                                                     (short)0, acc1, false, false);
      }
    }
    if (msk & 4) {
#pragma unroll
      for (int kk = 0; kk < 4; ++kk) {
        const float* p2 = wb + (size_t)(((2 * FTILES + ft) * 4 + kk) * 64) + lane * 2;
        v2f b2; b2.x = p2[0]; b2.y = p2[1];
        acc2 = __builtin_amdgcn_wmma_f32_16x16x4_f32(false, a2[kk], false, b2,
                                                     (short)0, acc2, false, false);
      }
    }
    __builtin_amdgcn_wave_barrier();   // keep next iter's LDS writes below reads
  }

  // ---- epilogue: 20*log10(fb==0 ? EPS : fb), bounds-checked store ----
  // (mel tiles whose W blocks were all skipped hold exact 0 -> EPS path,
  //  identical to the reference semantics)
#pragma unroll
  for (int r = 0; r < 8; ++r) {
    int fr = frbase + mofs + r;
    if (fr >= NFRAMES) continue;
    float* orow = out + ((size_t)b * NFRAMES + fr) * NFILT;
    {
      float x = acc0[r]; x = (x == 0.0f) ? EPSF : x;
      orow[col] = 20.0f * log10f(x);
    }
    {
      float x = acc1[r]; x = (x == 0.0f) ? EPSF : x;
      orow[16 + col] = 20.0f * log10f(x);
    }
    if (col < 8) {
      float x = acc2[r]; x = (x == 0.0f) ? EPSF : x;
      orow[32 + col] = 20.0f * log10f(x);
    }
  }
}

// ---------------------------------------------------------------------------
extern "C" void kernel_launch(void* const* d_in, const int* in_sizes, int n_in,
                              void* d_out, int out_size, void* d_ws, size_t ws_size,
                              hipStream_t stream) {
  const float* sig = (const float*)d_in[0];
  const int*   fsp = (const int*)d_in[1];
  _Float16* tw  = (_Float16*)d_ws;
  float*    wbp = (float*)((char*)d_ws + WB_OFF);
  int*      msk = (int*)((char*)d_ws + MASK_OFF);
  float*    out = (float*)d_out;

  twiddle_init<<<(TW_HALFS + 255) / 256, 256, 0, stream>>>(tw, msk);
  fbank_init<<<(WB_FLOATS + 255) / 256, 256, 0, stream>>>(wbp, msk, fsp);

  const int total_waves = NBATCH * FRTILES;          // 10016
  const int blocks = total_waves / WAVES_PB;         // 1252 (exact)
  mel_wmma<<<blocks, WAVES_PB * 32, 0, stream>>>(sig, tw, wbp, msk, out);
}
